// QuantMoETorchFFN_63522566308129
// MI455X (gfx1250) — matrix operand order repaired and checked by reading, not compiled, not run on hardware
//
#include <hip/hip_runtime.h>
#include <math.h>

// ---------------- problem constants ----------------
#define DIMD   2048
#define HID    5632
#define NEXP   8
#define NTOK   2048
#define NSLOT  (NTOK * 2)          // top-2 -> 4096 routed slots total

// ---------------- tiling ----------------
#define BM     128                 // tokens per block tile
#define BN     128                 // output cols per block tile
#define KC     32                  // K chunk (bf16 wmma K=32)
#define RSK    40                  // u16 stride per row-entity (32 k + pad; 80B = 16B aligned)

typedef __attribute__((ext_vector_type(16))) __bf16 v16bf;
typedef __attribute__((ext_vector_type(8)))  float  v8f;

union FragBF { unsigned int u[8]; v16bf v; };

// fp32 -> bf16 via native conversion (RNE)
__device__ __forceinline__ unsigned short f2bf(float f) {
    union { __bf16 b; unsigned short s; } c;
    c.b = (__bf16)f;
    return c.s;
}

__device__ __forceinline__ unsigned long long pack4bf(float a, float b, float c, float d) {
    return (unsigned long long)f2bf(a)
         | ((unsigned long long)f2bf(b) << 16)
         | ((unsigned long long)f2bf(c) << 32)
         | ((unsigned long long)f2bf(d) << 48);
}

__device__ __forceinline__ void load_frag(const unsigned short* p, FragBF& f) {
    uint4 lo = *(const uint4*)(p);        // 16B chunk 1
    uint4 hi = *(const uint4*)(p + 16);   // 16B chunk 2 (A: +16 u16; B layout contiguous too)
    f.u[0] = lo.x; f.u[1] = lo.y; f.u[2] = lo.z; f.u[3] = lo.w;
    f.u[4] = hi.x; f.u[5] = hi.y; f.u[6] = hi.z; f.u[7] = hi.w;
}
// B fragment: two 16B chunks at k offsets 0 and 16
__device__ __forceinline__ void load_fragB(const unsigned short* p, FragBF& f) {
    uint4 lo = *(const uint4*)(p);
    uint4 hi = *(const uint4*)(p + 16);
    f.u[0] = lo.x; f.u[1] = lo.y; f.u[2] = lo.z; f.u[3] = lo.w;
    f.u[4] = hi.x; f.u[5] = hi.y; f.u[6] = hi.z; f.u[7] = hi.w;
}

__device__ __forceinline__ v8f wmma_bf16(const FragBF& a, const FragBF& b, v8f c) {
    return __builtin_amdgcn_wmma_f32_16x16x32_bf16(false, a.v, false, b.v,
                                                   (short)0, c, false, false);
}

// ================= kernel 1: gating (1 wave per token) =================
__global__ __launch_bounds__(256) void k_gate(const float* __restrict__ x,
                                              const float* __restrict__ gw,
                                              int* __restrict__ eidx,
                                              float* __restrict__ ew) {
    const int lane = threadIdx.x & 31;
    const int wave = threadIdx.x >> 5;
    const int t = blockIdx.x * 8 + wave;
    float acc[NEXP];
#pragma unroll
    for (int e = 0; e < NEXP; ++e) acc[e] = 0.f;

    const float* xr = x + (size_t)t * DIMD;
#pragma unroll 4
    for (int d = lane; d < DIMD; d += 32) {
        float xv = xr[d];
#pragma unroll
        for (int e = 0; e < NEXP; ++e) acc[e] += xv * gw[e * DIMD + d];
    }
#pragma unroll
    for (int e = 0; e < NEXP; ++e)
#pragma unroll
        for (int off = 16; off > 0; off >>= 1)
            acc[e] += __shfl_xor(acc[e], off, 32);

    if (lane == 0) {
        int i1 = 0; float s1 = acc[0];
#pragma unroll
        for (int e = 1; e < NEXP; ++e) if (acc[e] > s1) { s1 = acc[e]; i1 = e; }
        int i2 = -1; float s2 = -3.4e38f;
#pragma unroll
        for (int e = 0; e < NEXP; ++e) if (e != i1 && acc[e] > s2) { s2 = acc[e]; i2 = e; }
        float e2 = expf(s2 - s1);          // s1 >= s2
        float inv = 1.f / (1.f + e2);
        eidx[t * 2 + 0] = i1; eidx[t * 2 + 1] = i2;
        ew[t * 2 + 0] = inv;  ew[t * 2 + 1] = e2 * inv;
    }
}

// ================= kernel 2: routing (single block counting sort) =================
__global__ __launch_bounds__(256) void k_route(const int* __restrict__ eidx,
                                               const float* __restrict__ ew,
                                               int* __restrict__ bases,
                                               int* __restrict__ rtok,
                                               float* __restrict__ rw) {
    __shared__ int cnt[NEXP];
    __shared__ int base_s[NEXP + 1];
    __shared__ int cur[NEXP];
    const int tid = threadIdx.x;
    if (tid < NEXP) cnt[tid] = 0;
    __syncthreads();
    for (int i = tid; i < NSLOT; i += 256) atomicAdd(&cnt[eidx[i]], 1);
    __syncthreads();
    if (tid == 0) {
        base_s[0] = 0;
        for (int e = 0; e < NEXP; ++e) base_s[e + 1] = base_s[e] + cnt[e];
    }
    __syncthreads();
    if (tid < NEXP) cur[tid] = base_s[tid];
    __syncthreads();
    for (int i = tid; i < NSLOT; i += 256) {
        int e = eidx[i];
        int p = atomicAdd(&cur[e], 1);
        rtok[p] = i >> 1;
        rw[p]   = ew[i];
    }
    __syncthreads();
    if (tid <= NEXP) bases[tid] = base_s[tid];
}

// ================= kernel 3: zero output (combine uses atomics) =================
__global__ __launch_bounds__(256) void k_zero(float* __restrict__ out, int n) {
    int i = blockIdx.x * 256 + threadIdx.x;
    if (i < n) out[i] = 0.f;
}

// ================= kernel 4: GEMM1 + SwiGLU -> h (bf16), double-buffered =================
// grid: (HID/BN, NTOK/BM, NEXP); block 256 (8 waves: 2 m-waves x 4 n-waves)
__global__ __launch_bounds__(256) void k_ffn1(const float* __restrict__ x,
                                              const float* __restrict__ w1,
                                              const float* __restrict__ w3,
                                              const int* __restrict__ bases,
                                              const int* __restrict__ rtok,
                                              unsigned short* __restrict__ hbuf) {
    const int e  = blockIdx.z;
    const int mt = blockIdx.y;
    const int nt = blockIdx.x;
    const int rb  = bases[e];
    const int cnt = bases[e + 1] - rb;
    if (mt * BM >= cnt) return;

    __shared__ __align__(16) unsigned short lA [2][BM * RSK];   // [m][k]
    __shared__ __align__(16) unsigned short lW1[2][BN * RSK];   // [n][k]
    __shared__ __align__(16) unsigned short lW3[2][BN * RSK];   // [n][k]

    const int tid  = threadIdx.x;
    const int lane = tid & 31;
    const int wave = tid >> 5;
    const int wm = wave & 1;     // 0..1  -> 64-row strip
    const int wn = wave >> 1;    // 0..3  -> 32-col strip

    // ---- hoisted per-thread staging state (constant across K loop) ----
    const int k4  = tid & 7;     // A: float4 slot within 32-k chunk
    const int rA0 = tid >> 3;    // A: rows rA0 + 32*s
    const float* xrow[4];
    int aoff[4];
#pragma unroll
    for (int s = 0; s < 4; ++s) {
        int r  = rA0 + 32 * s;
        int gr = mt * BM + r;
        int t  = rtok[rb + (gr < cnt ? gr : cnt - 1)];  // clamp: garbage rows discarded later
        xrow[s] = x + (size_t)t * DIMD + k4 * 4;
        aoff[s] = r * RSK + k4 * 4;
    }
    const int kW = tid >> 5;     // W: k row (kW + 8*s)
    const int n4 = tid & 31;     // W: float4 slot over n
    const float* w1q = w1 + (size_t)e * DIMD * HID + (size_t)nt * BN + (size_t)kW * HID + n4 * 4;
    const float* w3q = w3 + (size_t)e * DIMD * HID + (size_t)nt * BN + (size_t)kW * HID + n4 * 4;
    int woff[4];
#pragma unroll
    for (int i = 0; i < 4; ++i) woff[i] = (n4 * 4 + i) * RSK + kW;

    auto stage = [&](int b) {
#pragma unroll
        for (int s = 0; s < 4; ++s) {
            float4 v = *(const float4*)xrow[s];
            xrow[s] += KC;
            *(unsigned long long*)&lA[b][aoff[s]] = pack4bf(v.x, v.y, v.z, v.w);
        }
#pragma unroll
        for (int s = 0; s < 4; ++s) {
            float4 g1 = *(const float4*)(w1q + (size_t)(8 * s) * HID);
            float4 g3 = *(const float4*)(w3q + (size_t)(8 * s) * HID);
            lW1[b][woff[0] + 8 * s] = f2bf(g1.x);
            lW1[b][woff[1] + 8 * s] = f2bf(g1.y);
            lW1[b][woff[2] + 8 * s] = f2bf(g1.z);
            lW1[b][woff[3] + 8 * s] = f2bf(g1.w);
            lW3[b][woff[0] + 8 * s] = f2bf(g3.x);
            lW3[b][woff[1] + 8 * s] = f2bf(g3.y);
            lW3[b][woff[2] + 8 * s] = f2bf(g3.z);
            lW3[b][woff[3] + 8 * s] = f2bf(g3.w);
        }
        __builtin_prefetch(w1q + (size_t)KC * HID, 0, 1);
        __builtin_prefetch(w3q + (size_t)KC * HID, 0, 1);
        w1q += (size_t)KC * HID;
        w3q += (size_t)KC * HID;
    };

    // ---- fragment LDS offsets (u16 units), constant per lane ----
    const int halfL = lane >> 4, low = lane & 15;
    int aF[4], bF[2];
#pragma unroll
    for (int i = 0; i < 4; ++i) aF[i] = (wm * 64 + i * 16 + low) * RSK + 8 * halfL;
#pragma unroll
    for (int j = 0; j < 2; ++j) bF[j] = (wn * 32 + j * 16 + low) * RSK + 8 * halfL;

    v8f acc1[4][2], acc3[4][2];
    const v8f zf = {0.f, 0.f, 0.f, 0.f, 0.f, 0.f, 0.f, 0.f};
#pragma unroll
    for (int i = 0; i < 4; ++i)
#pragma unroll
        for (int j = 0; j < 2; ++j) { acc1[i][j] = zf; acc3[i][j] = zf; }

    const int KCH = DIMD / KC;   // 64
    stage(0);
    __syncthreads();
    int p = 0;
    for (int kc = 0; kc < KCH; ++kc) {
        if (kc + 1 < KCH) stage(p ^ 1);   // overlap next-chunk staging with WMMA

        FragBF fa[4], f1[2], f3[2];
#pragma unroll
        for (int i = 0; i < 4; ++i) load_frag(&lA[p][aF[i]], fa[i]);
#pragma unroll
        for (int j = 0; j < 2; ++j) {
            load_fragB(&lW1[p][bF[j]], f1[j]);
            load_fragB(&lW3[p][bF[j]], f3[j]);
        }
#pragma unroll
        for (int i = 0; i < 4; ++i)
#pragma unroll
            for (int j = 0; j < 2; ++j) {
                acc1[i][j] = wmma_bf16(fa[i], f1[j], acc1[i][j]);
                acc3[i][j] = wmma_bf16(fa[i], f3[j], acc3[i][j]);
            }
        __syncthreads();
        p ^= 1;
    }

    // epilogue: h = silu(a1) * a3 -> bf16
#pragma unroll
    for (int i = 0; i < 4; ++i)
#pragma unroll
        for (int j = 0; j < 2; ++j)
#pragma unroll
            for (int v = 0; v < 8; ++v) {
                int msub = v + 8 * halfL;
                int gm = mt * BM + wm * 64 + i * 16 + msub;
                if (gm < cnt) {
                    float g = acc1[i][j][v];
                    float hv = (g / (1.f + expf(-g))) * acc3[i][j][v];
                    int gn = nt * BN + wn * 32 + j * 16 + low;
                    hbuf[(size_t)(rb + gm) * HID + gn] = f2bf(hv);
                }
            }
}

// ================= kernel 5: GEMM2 + weighted combine, async-DMA A, double-buffered =================
// grid: (DIMD/BN, NTOK/BM, NEXP); block 256
__global__ __launch_bounds__(256) void k_ffn2(const unsigned short* __restrict__ hbuf,
                                              const float* __restrict__ w2,
                                              const int* __restrict__ bases,
                                              const int* __restrict__ rtok,
                                              const float* __restrict__ rw,
                                              float* __restrict__ out) {
    const int e  = blockIdx.z;
    const int mt = blockIdx.y;
    const int nt = blockIdx.x;
    const int rb  = bases[e];
    const int cnt = bases[e + 1] - rb;
    if (mt * BM >= cnt) return;

    __shared__ __align__(16) unsigned short lA[2][BM * RSK];    // [m][k]
    __shared__ __align__(16) unsigned short lW[2][BN * RSK];    // [n][k]
    __shared__ int   tokS[BM];
    __shared__ float wS[BM];

    const int tid  = threadIdx.x;
    const int lane = tid & 31;
    const int wave = tid >> 5;
    const int wm = wave & 1;
    const int wn = wave >> 1;

    if (tid < BM) {
        int gr = mt * BM + tid;
        bool ok = gr < cnt;
        tokS[tid] = ok ? rtok[rb + gr] : -1;
        wS[tid]   = ok ? rw[rb + gr]   : 0.f;
    }

    // ---- hoisted per-thread staging state ----
    const int k4  = tid & 7;
    const int rA0 = tid >> 3;
    const unsigned short* hq[4];
    unsigned int lsoA[2][4];                 // LDS byte addresses per buffer
#pragma unroll
    for (int s = 0; s < 4; ++s) {
        int r   = rA0 + 32 * s;
        int row = rb + mt * BM + r;
        if (row > NSLOT - 1) row = NSLOT - 1;   // clamp: garbage rows discarded at epilogue
        hq[s] = hbuf + (size_t)row * HID + k4 * 4;
        int ao = r * RSK + k4 * 4;
        lsoA[0][s] = (unsigned int)(size_t)&lA[0][ao];
        lsoA[1][s] = (unsigned int)(size_t)&lA[1][ao];
    }
    const int kW = tid >> 5;
    const int n4 = tid & 31;
    const float* w2q = w2 + (size_t)e * HID * DIMD + (size_t)nt * BN + (size_t)kW * DIMD + n4 * 4;
    int woff[4];
#pragma unroll
    for (int i = 0; i < 4; ++i) woff[i] = (n4 * 4 + i) * RSK + kW;

    auto stage = [&](int b) {
#pragma unroll
        for (int s = 0; s < 4; ++s) {      // async DMA: global (bf16) -> LDS, 8B each
            asm volatile("global_load_async_to_lds_b64 %0, %1, off"
                         :: "v"(lsoA[b][s]), "v"((unsigned long long)(size_t)hq[s])
                         : "memory");
            hq[s] += KC;
        }
#pragma unroll
        for (int s = 0; s < 4; ++s) {
            float4 g = *(const float4*)(w2q + (size_t)(8 * s) * DIMD);
            lW[b][woff[0] + 8 * s] = f2bf(g.x);
            lW[b][woff[1] + 8 * s] = f2bf(g.y);
            lW[b][woff[2] + 8 * s] = f2bf(g.z);
            lW[b][woff[3] + 8 * s] = f2bf(g.w);
        }
        __builtin_prefetch(w2q + (size_t)KC * DIMD, 0, 1);
        w2q += (size_t)KC * DIMD;
    };

    const int halfL = lane >> 4, low = lane & 15;
    int aF[4], bF[2];
#pragma unroll
    for (int i = 0; i < 4; ++i) aF[i] = (wm * 64 + i * 16 + low) * RSK + 8 * halfL;
#pragma unroll
    for (int j = 0; j < 2; ++j) bF[j] = (wn * 32 + j * 16 + low) * RSK + 8 * halfL;

    v8f acc[4][2];
    const v8f zf = {0.f, 0.f, 0.f, 0.f, 0.f, 0.f, 0.f, 0.f};
#pragma unroll
    for (int i = 0; i < 4; ++i)
#pragma unroll
        for (int j = 0; j < 2; ++j) acc[i][j] = zf;

    const int KCH = HID / KC;    // 176
    stage(0);
    asm volatile("s_wait_asynccnt 0x0" ::: "memory");
    __syncthreads();
    int p = 0;
    for (int kc = 0; kc < KCH; ++kc) {
        if (kc + 1 < KCH) stage(p ^ 1);    // DMA + W staging run under the WMMAs

        FragBF fa[4], fb[2];
#pragma unroll
        for (int i = 0; i < 4; ++i) load_frag(&lA[p][aF[i]], fa[i]);
#pragma unroll
        for (int j = 0; j < 2; ++j) load_fragB(&lW[p][bF[j]], fb[j]);
#pragma unroll
        for (int i = 0; i < 4; ++i)
#pragma unroll
            for (int j = 0; j < 2; ++j)
                acc[i][j] = wmma_bf16(fa[i], fb[j], acc[i][j]);

        asm volatile("s_wait_asynccnt 0x0" ::: "memory");
        __syncthreads();
        p ^= 1;
    }

    // epilogue: out[token, n] += cw * y   (exactly 2 contributions per token -> commutative)
#pragma unroll
    for (int i = 0; i < 4; ++i)
#pragma unroll
        for (int j = 0; j < 2; ++j)
#pragma unroll
            for (int v = 0; v < 8; ++v) {
                int lm = wm * 64 + i * 16 + v + 8 * halfL;
                int t = tokS[lm];
                if (t >= 0) {
                    int gn = nt * BN + wn * 32 + j * 16 + low;
                    atomicAdd(&out[(size_t)t * DIMD + gn], acc[i][j][v] * wS[lm]);
                }
            }
}

// ================= host-side launch =================
extern "C" void kernel_launch(void* const* d_in, const int* in_sizes, int n_in,
                              void* d_out, int out_size, void* d_ws, size_t ws_size,
                              hipStream_t stream) {
    const float* x  = (const float*)d_in[0];
    const float* gw = (const float*)d_in[1];
    const float* w1 = (const float*)d_in[2];
    const float* w2 = (const float*)d_in[3];
    const float* w3 = (const float*)d_in[4];
    float* out = (float*)d_out;

    // workspace carve-up (256B aligned)
    char* w = (char*)d_ws;
    size_t off = 0;
    auto carve = [&](size_t bytes) { void* p = w + off; off = (off + bytes + 255) & ~(size_t)255; return p; };
    int*            eidx  = (int*)           carve(NSLOT * sizeof(int));
    float*          ew    = (float*)         carve(NSLOT * sizeof(float));
    int*            bases = (int*)           carve(16 * sizeof(int));
    int*            rtok  = (int*)           carve(NSLOT * sizeof(int));
    float*          rw    = (float*)         carve(NSLOT * sizeof(float));
    unsigned short* hbuf  = (unsigned short*)carve((size_t)NSLOT * HID * sizeof(unsigned short));
    (void)ws_size; (void)in_sizes; (void)n_in; (void)out_size;

    k_gate <<<NTOK / 8, 256, 0, stream>>>(x, gw, eidx, ew);
    k_route<<<1, 256, 0, stream>>>(eidx, ew, bases, rtok, rw);
    k_zero <<<(NTOK * DIMD + 255) / 256, 256, 0, stream>>>(out, NTOK * DIMD);
    k_ffn1 <<<dim3(HID / BN, NTOK / BM, NEXP), 256, 0, stream>>>(x, w1, w3, bases, rtok, hbuf);
    k_ffn2 <<<dim3(DIMD / BN, NTOK / BM, NEXP), 256, 0, stream>>>(hbuf, w2, bases, rtok, rw, out);
}
